// E_GCL_vel_2241972928558
// MI455X (gfx1250) — matrix operand order repaired and checked by reading, not compiled, not run on hardware
//
#include <hip/hip_runtime.h>

typedef __attribute__((ext_vector_type(16))) __bf16          v16bf;
typedef __attribute__((ext_vector_type(16))) unsigned short  v16us;
typedef __attribute__((ext_vector_type(8)))  float           v8f;

#define GROUP     5
#define H         64
#define GPB       2                   // graphs per block
#define NODES_PB  (GPB * GROUP)       // 10
#define TUPLES_PB (GPB * 25)          // 50
#define EDGES_PB  (GPB * 20)          // 40
#define WL_ROWS   64                  // 50 padded to 64 (4 M-tiles)
#define ED_ROWS   48                  // 40 padded to 48 (3 M-tiles)
#define THREADS   128
#define NWAVES    4

__device__ __forceinline__ unsigned short f2bf(float f) {
  unsigned u = __builtin_bit_cast(unsigned, f);
  u += 0x7FFFu + ((u >> 16) & 1u);            // round-to-nearest-even
  return (unsigned short)(u >> 16);
}
__device__ __forceinline__ float bf2f(unsigned short b) {
  unsigned u = ((unsigned)b) << 16;
  return __builtin_bit_cast(float, u);
}

// D(bf16, lds) = [res +] (relu)( A(bf16 lds, lda) x W(f32 global, [Kr x 64]) + bias )
// Wave w owns N-tile nt==w; B fragments loaded ONCE into registers, reused
// across all MT row-tiles.  KT = number of 16x16x32 k-steps (compile-time).
template <int KT>
__device__ __forceinline__ void gemm64(const unsigned short* A, int lda, int MT,
                                       const float* W, int Kr, const float* bias,
                                       bool relu, unsigned short* D,
                                       const unsigned short* resbf,
                                       int wave, int lane) {
  const int nt    = wave;                    // NWAVES == 4 == N-tiles
  const int ncol  = nt * 16 + (lane & 15);
  const int khalf = lane >> 4;

  v16us bu[KT];
#pragma unroll
  for (int ks = 0; ks < KT; ++ks) {
#pragma unroll
    for (int e = 0; e < 16; ++e) {
      // B 32x16 bf16 layout: lanes 0-15 -> K {0..7,16..23}, lanes 16-31 -> {8..15,24..31}
      int kb = ks * 32 + (e & 7) + ((e >> 3) << 4) + khalf * 8;
      bu[ks][e] = (kb < Kr) ? f2bf(W[kb * H + ncol]) : (unsigned short)0;
    }
  }

  for (int mt = 0; mt < MT; ++mt) {
    const unsigned short* arow = A + (mt * 16 + (lane & 15)) * lda;
    v8f c = {};
#pragma unroll
    for (int ks = 0; ks < KT; ++ks) {
      v16us au;
#pragma unroll
      for (int e = 0; e < 16; ++e) {
        // A 16x32 bf16 layout: lanes 0-15 -> K {0..7,16..23}, lanes 16-31 -> {8..15,24..31}
        au[e] = arow[ks * 32 + (e & 7) + ((e >> 3) << 4) + khalf * 8];
      }
      c = __builtin_amdgcn_wmma_f32_16x16x32_bf16(
          false, __builtin_bit_cast(v16bf, au),
          false, __builtin_bit_cast(v16bf, bu[ks]),
          (short)0, c, false, false);
    }
    const float b  = bias ? bias[ncol] : 0.f;
    const int   mb = mt * 16 + khalf * 8;
#pragma unroll
    for (int r = 0; r < 8; ++r) {
      float v = c[r] + b;
      if (relu) v = fmaxf(v, 0.f);
      if (resbf) v += bf2f(resbf[(mb + r) * H + ncol]);
      D[(mb + r) * H + ncol] = f2bf(v);
    }
  }
}

// ExpNormal RBF (3 centers) with cosine cutoff + fixed normalization
__device__ __forceinline__ void rbf3(float dist, float* out3) {
  const float start = 4.53999298e-05f;                   // exp(-10)
  const float bx = (2.0f / 3.0f) * (1.0f - start);
  const float beta = 1.0f / (bx * bx);
  const float means0 = start, means1 = 0.5f * (1.0f + start), means2 = 1.0f;
  float cutoff = (dist <= 10.0f) ? 0.5f * (__cosf(dist * 0.31415926535f) + 1.0f) : 0.0f;
  float ex = __expf(-0.5f * dist);
  float d0 = ex - means0, d1 = ex - means1, d2 = ex - means2;
  out3[0] = (cutoff * __expf(-beta * d0 * d0) - 0.05f) * (1.0f / 0.15f);
  out3[1] = (cutoff * __expf(-beta * d1 * d1) - 0.05f) * (1.0f / 0.15f);
  out3[2] = (cutoff * __expf(-beta * d2 * d2) - 0.05f) * (1.0f / 0.15f);
}

struct Params {
  const float *h, *coord, *vel;
  const float *e_w1, *e_b1, *e_w2, *e_b2;
  const float *n_w1, *n_b1, *n_w2, *n_b2;
  const float *c_w1, *c_b1, *c_w2;
  const float *v_w1, *v_b1, *v_w2, *v_b2;
  const float *ic_wef, *ic_pat;
  const float *br_w1, *br_b1, *br_w2, *br_b2;
  const float *out_w, *out_b;
  float *h_out, *coord_out;
  int n_nodes;
};

__global__ __launch_bounds__(THREADS) void egcl_fused(Params p) {
  __shared__ unsigned short s_arena[5 * WL_ROWS * H];   // 5 aliasable 64x64 bf16 buffers
  __shared__ float s_h[16 * H];                          // h rows, padded to 16
  __shared__ float s_c[NODES_PB * 3], s_v[NODES_PB * 3];
  __shared__ float s_ef[TUPLES_PB * 12];
  __shared__ float s_rad[EDGES_PB], s_cmat[EDGES_PB], s_vmat[NODES_PB];

  unsigned short* KEMB = s_arena + 0 * WL_ROWS * H;
  unsigned short* SMB  = s_arena + 1 * WL_ROWS * H;
  unsigned short* K0B  = s_arena + 2 * WL_ROWS * H;
  unsigned short* K1B  = s_arena + 3 * WL_ROWS * H;
  unsigned short* TMP  = s_arena + 4 * WL_ROWS * H;
  unsigned short* XE   = SMB;            // 48*224 bf16, spans SMB..K1B (dead by then)
  unsigned short* HBF  = SMB + 10752;    // bf16 copy of h (16x64), after XE
  unsigned short* VH   = SMB;            // vel-head hidden (16x64), XE dead
  unsigned short* EF   = KEMB;           // edge_feat 48*64 (kemb dead after XE built)
  unsigned short* X2   = SMB;            // 16*128 node-model input (VH dead)

  const int tid  = threadIdx.x;
  const int lane = tid & 31, wave = tid >> 5;
  const int g0    = blockIdx.x * GPB;
  const int nbase = g0 * GROUP;

  // warm L2 for the big weight panels (global_prefetch_b8)
  if (tid == 0) {
    __builtin_prefetch(p.e_w1, 0, 1);
    __builtin_prefetch(p.br_w1, 0, 1);
    __builtin_prefetch(p.br_w2, 0, 1);
    __builtin_prefetch(p.n_w1, 0, 1);
    __builtin_prefetch(p.out_w, 0, 1);
  }

  // ---- stage 1: load h / coord / vel into LDS (zero-pad rows) ----
  for (int idx = tid; idx < 16 * H; idx += THREADS) {
    int n = idx >> 6;
    s_h[idx] = (n < NODES_PB && (nbase + n) < p.n_nodes)
                   ? p.h[(size_t)(nbase + n) * H + (idx & 63)] : 0.f;
  }
  for (int idx = tid; idx < NODES_PB * 3; idx += THREADS) {
    bool ok = (size_t)nbase * 3 + idx < (size_t)p.n_nodes * 3;
    s_c[idx] = ok ? p.coord[(size_t)nbase * 3 + idx] : 0.f;
    s_v[idx] = ok ? p.vel[(size_t)nbase * 3 + idx]   : 0.f;
  }
  __syncthreads();

  // ---- stage 2: edge features ef (50 tuples x 12) + per-edge radial ----
  for (int w = tid; w < TUPLES_PB * 4; w += THREADS) {
    int t = w >> 2, grp = w & 3;
    int g = t / 25, r = t % 25, i = r / 5, j = r % 5;
    int ni = (g * GROUP + i) * 3, nj = (g * GROUP + j) * 3;
    const float *A3, *B3;
    if (grp == 0)      { A3 = s_c + ni; B3 = s_c + nj; }   // ||c_i - c_j||
    else if (grp == 1) { A3 = s_v + ni; B3 = s_v + nj; }   // ||v_i - v_j||
    else if (grp == 2) { A3 = s_c + ni; B3 = s_v + nj; }   // mcd[i][j]
    else               { A3 = s_c + nj; B3 = s_v + ni; }   // mcd^T[i][j]
    float dx = A3[0] - B3[0], dy = A3[1] - B3[1], dz = A3[2] - B3[2];
    float sq = dx * dx + dy * dy + dz * dz;
    float dist = (sq > 0.f) ? sqrtf(sq) : 0.f;
    rbf3(dist, &s_ef[t * 12 + grp * 3]);
  }
  for (int e = tid; e < EDGES_PB; e += THREADS) {
    int g = e / 20, pe = e % 20, i = pe >> 2, q = pe & 3, j = q + (q >= i);
    int ni = (g * GROUP + i) * 3, nj = (g * GROUP + j) * 3;
    float dx = s_c[ni] - s_c[nj], dy = s_c[ni + 1] - s_c[nj + 1], dz = s_c[ni + 2] - s_c[nj + 2];
    s_rad[e] = dx * dx + dy * dy + dz * dz;                 // radial = |coord_diff|^2
  }
  __syncthreads();

  // ---- stage 3: kemb0 = (ef @ ic_wef) * ic_pat[pattern]  -> KEMB (bf16) ----
  for (int idx = tid; idx < WL_ROWS * H; idx += THREADS) {
    int row = idx >> 6, d = idx & 63;
    float v = 0.f;
    if (row < TUPLES_PB) {
      int r = row % 25, i = r / 5, j = r % 5;
      float acc = 0.f;
#pragma unroll
      for (int e = 0; e < 12; ++e) acc += s_ef[row * 12 + e] * p.ic_wef[e * H + d];
      v = acc * p.ic_pat[((i == j) ? 2 : 1) * H + d];
    }
    KEMB[idx] = f2bf(v);
  }
  __syncthreads();

  // ---- stage 4: three branch MLPs (WMMA), sm/k0/k1 ----
  unsigned short* dsts[3] = {SMB, K0B, K1B};
  for (int br = 0; br < 3; ++br) {
    gemm64<2>(KEMB, H, 4, p.br_w1 + br * H * H, H, p.br_b1 + br * H, true, TMP, nullptr, wave, lane);
    __syncthreads();
    gemm64<2>(TMP, H, 4, p.br_w2 + br * H * H, H, p.br_b2 + br * H, true, dsts[br], nullptr, wave, lane);
    __syncthreads();
  }

  // ---- stage 5: per-channel 5x5x5 einsum, smmult = sm * (k0 @ k1) -> TMP ----
  for (int idx = tid; idx < TUPLES_PB * H; idx += THREADS) {
    int row = idx >> 6, d = idx & 63;
    int g = row / 25, r = row % 25, i = r / 5, k = r % 5;
    float acc = 0.f;
#pragma unroll
    for (int j = 0; j < 5; ++j)
      acc += bf2f(K0B[(g * 25 + i * 5 + j) * H + d]) * bf2f(K1B[(g * 25 + j * 5 + k) * H + d]);
    TMP[row * H + d] = f2bf(bf2f(SMB[row * H + d]) * acc);
  }
  __syncthreads();

  // ---- stage 6: kemb += relu(smmult @ out_w + out_b)  (residual in-place) ----
  gemm64<2>(TMP, H, 4, p.out_w, H, p.out_b, true, KEMB, KEMB, wave, lane);
  __syncthreads();

  // ---- stage 7: edge input X = [h_i | h_j | radial | kemb_ij | 0] (48x224)
  //               + bf16 copy of h (HBF, 16x64) ----
  for (int idx = tid; idx < ED_ROWS * 224; idx += THREADS) {
    int e = idx / 224, col = idx % 224;
    unsigned short v = 0;
    if (e < EDGES_PB) {
      int g = e / 20, pe = e % 20, i = pe >> 2, q = pe & 3, j = q + (q >= i);
      if (col < 64)        v = f2bf(s_h[(g * GROUP + i) * H + col]);
      else if (col < 128)  v = f2bf(s_h[(g * GROUP + j) * H + (col - 64)]);
      else if (col == 128) v = f2bf(s_rad[e]);
      else if (col < 193)  v = KEMB[(g * 25 + i * 5 + j) * H + (col - 129)];
    }
    XE[idx] = v;
  }
  for (int idx = tid; idx < 16 * H; idx += THREADS)
    HBF[idx] = ((idx >> 6) < NODES_PB) ? f2bf(s_h[idx]) : (unsigned short)0;
  __syncthreads();

  // ---- stage 8: edge MLP (193->64 relu, 64->64 relu) ----
  gemm64<7>(XE, 224, 3, p.e_w1, 193, p.e_b1, true, TMP, nullptr, wave, lane);
  __syncthreads();
  gemm64<2>(TMP, H, 3, p.e_w2, H, p.e_b2, true, EF, nullptr, wave, lane);
  __syncthreads();

  // ---- stage 9: coord head hidden = relu(edge_feat @ c_w1 + c_b1) -> TMP ----
  gemm64<2>(EF, H, 3, p.c_w1, H, p.c_b1, true, TMP, nullptr, wave, lane);
  __syncthreads();

  // ---- stage 10a: vel head hidden = relu(h @ v_w1 + v_b1) -> VH (WMMA) ----
  gemm64<2>(HBF, H, 1, p.v_w1, H, p.v_b1, true, VH, nullptr, wave, lane);
  __syncthreads();

  // ---- stage 10b: cmat per edge + vmat per node (64-wide reductions) ----
  for (int e = tid; e < EDGES_PB; e += THREADS) {
    float acc = 0.f;
    for (int d = 0; d < H; ++d) acc += bf2f(TMP[e * H + d]) * p.c_w2[d];
    s_cmat[e] = acc;
  }
  if (tid < NODES_PB) {
    float acc = 0.f;
    for (int d = 0; d < H; ++d) acc += bf2f(VH[tid * H + d]) * p.v_w2[d];
    s_vmat[tid] = acc + p.v_b2[0];
  }
  __syncthreads();

  // ---- stage 11: coord_out = coord + mean(clip(diff*cmat)) + vel*vmat ----
  if (tid < NODES_PB * 3) {
    int n = tid / 3, dim = tid % 3;
    int g = n / GROUP, i = n % GROUP;
    float acc = 0.f;
#pragma unroll
    for (int q = 0; q < 4; ++q) {
      int j = q + (q >= i);
      float diff = s_c[n * 3 + dim] - s_c[(g * GROUP + j) * 3 + dim];
      float tr = diff * s_cmat[g * 20 + i * 4 + q];
      acc += fminf(fmaxf(tr, -100.f), 100.f);
    }
    if (nbase + n < p.n_nodes)
      p.coord_out[(size_t)(nbase + n) * 3 + dim] =
          s_c[n * 3 + dim] + acc * 0.25f + s_v[n * 3 + dim] * s_vmat[n];
  }

  // ---- stage 12: node input X2 = [h | nagg] (16x128) ----
  for (int idx = tid; idx < 16 * 128; idx += THREADS) {
    int n = idx >> 7, col = idx & 127;
    unsigned short v = 0;
    if (n < NODES_PB) {
      if (col < 64) v = HBF[n * H + col];
      else {
        int d = col - 64, g = n / GROUP, i = n % GROUP;
        float acc = 0.f;
#pragma unroll
        for (int q = 0; q < 4; ++q) acc += bf2f(EF[(g * 20 + i * 4 + q) * H + d]);
        v = f2bf(acc);
      }
    }
    X2[idx] = v;
  }
  __syncthreads();

  // ---- stage 13: node MLP hidden = relu(X2 @ n_w1 + n_b1) -> TMP ----
  gemm64<4>(X2, 128, 1, p.n_w1, 128, p.n_b1, true, TMP, nullptr, wave, lane);
  __syncthreads();

  // ---- stage 14: h_out = h + hidden @ n_w2 + n_b2  (f32, straight to global) ----
  {
    const int nt    = wave;
    const int ncol  = nt * 16 + (lane & 15);
    const int khalf = lane >> 4;
    v8f c = {};
#pragma unroll
    for (int ks = 0; ks < 2; ++ks) {
      v16us au, bu;
#pragma unroll
      for (int e = 0; e < 16; ++e) {
        au[e] = TMP[(lane & 15) * H + ks * 32 + (e & 7) + ((e >> 3) << 4) + khalf * 8];
        bu[e] = f2bf(p.n_w2[(ks * 32 + (e & 7) + ((e >> 3) << 4) + khalf * 8) * H + ncol]);
      }
      c = __builtin_amdgcn_wmma_f32_16x16x32_bf16(
          false, __builtin_bit_cast(v16bf, au),
          false, __builtin_bit_cast(v16bf, bu),
          (short)0, c, false, false);
    }
    const int mb = khalf * 8;
#pragma unroll
    for (int r = 0; r < 8; ++r) {
      int m = mb + r;
      if (m < NODES_PB && (nbase + m) < p.n_nodes)
        p.h_out[(size_t)(nbase + m) * H + ncol] = c[r] + p.n_b2[ncol] + s_h[m * H + ncol];
    }
  }
}

extern "C" void kernel_launch(void* const* d_in, const int* in_sizes, int n_in,
                              void* d_out, int out_size, void* d_ws, size_t ws_size,
                              hipStream_t stream) {
  (void)n_in; (void)out_size; (void)d_ws; (void)ws_size;
  Params p;
  p.h     = (const float*)d_in[0];
  p.coord = (const float*)d_in[1];
  p.vel   = (const float*)d_in[2];
  /* d_in[3] = edge_index (structure regenerated analytically) */
  p.e_w1 = (const float*)d_in[4];  p.e_b1 = (const float*)d_in[5];
  p.e_w2 = (const float*)d_in[6];  p.e_b2 = (const float*)d_in[7];
  p.n_w1 = (const float*)d_in[8];  p.n_b1 = (const float*)d_in[9];
  p.n_w2 = (const float*)d_in[10]; p.n_b2 = (const float*)d_in[11];
  p.c_w1 = (const float*)d_in[12]; p.c_b1 = (const float*)d_in[13];
  p.c_w2 = (const float*)d_in[14];
  p.v_w1 = (const float*)d_in[15]; p.v_b1 = (const float*)d_in[16];
  p.v_w2 = (const float*)d_in[17]; p.v_b2 = (const float*)d_in[18];
  p.ic_wef = (const float*)d_in[19]; p.ic_pat = (const float*)d_in[20];
  p.br_w1 = (const float*)d_in[21]; p.br_b1 = (const float*)d_in[22];
  p.br_w2 = (const float*)d_in[23]; p.br_b2 = (const float*)d_in[24];
  p.out_w = (const float*)d_in[25]; p.out_b = (const float*)d_in[26];

  const int N = in_sizes[0] / H;
  p.n_nodes  = N;
  p.h_out    = (float*)d_out;
  p.coord_out = (float*)d_out + (size_t)N * H;

  const int nGraphs = N / GROUP;
  const int blocks  = (nGraphs + GPB - 1) / GPB;
  egcl_fused<<<blocks, THREADS, 0, stream>>>(p);
}
